// fusion_layer_40836549050961
// MI455X (gfx1250) — matrix operand order repaired and checked by reading, not compiled, not run on hardware
//
#include <hip/hip_runtime.h>
#include <hip/hip_bf16.h>

// ---------------------------------------------------------------------------
// MI455X (gfx1250, wave32). bf16 operands end-to-end; GEMMs run
// v_wmma_f32_16x16x32_bf16. The A/B^T GEMM path stages tiles with the Tensor
// Data Mover in a 2-deep double-buffered pipeline (tensor_load_to_lds of tile
// i+1 overlaps WMMA of tile i; s_wait_tensorcnt 2 retires only tile i).
// ---------------------------------------------------------------------------

typedef __bf16 bf16;
typedef __attribute__((ext_vector_type(16))) __bf16 v16bf;
typedef __attribute__((ext_vector_type(8)))  __bf16 v8bf;
typedef __attribute__((ext_vector_type(8)))  float  v8f;
typedef __attribute__((ext_vector_type(4)))  unsigned int v4u;
typedef __attribute__((ext_vector_type(4)))  int v4i_t;
typedef __attribute__((ext_vector_type(8)))  int v8i_t;

#define HH   1024   // hidden
#define TEE  128    // t_embed
#define NB   8      // batch
#define NT   512    // T
#define NS   512    // S
#define K3   384    // 3*TE
#define BT   4096   // B*T == B*S

#if defined(__has_builtin)
#  if __has_builtin(__builtin_amdgcn_tensor_load_to_lds)
#    define HAS_TDM 1
#  endif
#endif
#ifndef HAS_TDM
#  define HAS_TDM 0
#endif

__device__ __forceinline__ bf16 f2bf(float f) {
    union { float f; unsigned u; } c; c.f = f;
    unsigned r = (c.u + 0x7FFFu + ((c.u >> 16) & 1u)) >> 16;
    union { unsigned short s; bf16 b; } o; o.s = (unsigned short)r;
    return o.b;
}
__device__ __forceinline__ float bf2f(bf16 b) {
    union { bf16 b; unsigned short s; } i; i.b = b;
    union { unsigned u; float f; } c; c.u = (unsigned)i.s << 16;
    return c.f;
}

#if HAS_TDM
typedef __attribute__((address_space(3))) void lds_void_t;
__device__ __forceinline__ unsigned lds_off_u32(const void* p) {
    return (unsigned)(unsigned long long)(const lds_void_t*)p;
}
// 2D TDM tile load: tile_k contiguous bf16 elements per row, `rows` rows,
// source row stride ld (elements). LDS rows padded: pad_interval code 3
// (every 16 DWORDs = one 32-half row), pad_amount code 3 (4 DWORDs = 8
// halves) -> matches the LDS_S = 40-half LDS row stride.
__device__ __forceinline__ void tdm_load_tile(unsigned lds_addr, const bf16* g,
                                              unsigned tile_k, unsigned rows,
                                              unsigned ld) {
    unsigned long long ga = (unsigned long long)g;
    v4u g0;
    g0[0] = 1u;                                   // count=1 (valid user D#)
    g0[1] = lds_addr;                             // lds_addr
    g0[2] = (unsigned)(ga & 0xFFFFFFFFu);         // global_addr[31:0]
    g0[3] = (unsigned)((ga >> 32) & 0x01FFFFFFu)  // global_addr[56:32]
          | (2u << 30);                           // type=2 ("image")
    v8i_t g1;
    g1[0] = (int)((1u << 16)        // data_size = 1 (2 bytes)
                | (1u << 20)        // pad_enable
                | (3u << 22)        // pad_interval: 16 DWORDs
                | (3u << 25));      // pad_amount:   4 DWORDs
    g1[1] = (int)(tile_k << 16);                  // tensor_dim0[15:0]
    g1[2] = (int)((tile_k >> 16) | (rows << 16)); // dim0 hi | tensor_dim1 lo
    g1[3] = (int)((rows >> 16) | (tile_k << 16)); // dim1 hi | tile_dim0
    g1[4] = (int)(rows & 0xFFFFu);                // tile_dim1 (tile_dim2=0)
    g1[5] = (int)ld;                              // tensor_dim0_stride[31:0]
    g1[6] = 0;
    g1[7] = 0;
    v4i_t z4 = {0, 0, 0, 0};
#if __clang_major__ >= 23
    v8i_t z8 = {0, 0, 0, 0, 0, 0, 0, 0};
    __builtin_amdgcn_tensor_load_to_lds(g0, g1, z4, z4, z8, 0);
#else
    __builtin_amdgcn_tensor_load_to_lds(g0, g1, z4, z4, 0);
#endif
}
#endif  // HAS_TDM

// Fragment loads (ISA 16-bit WMMA layouts) + 2x4 wmma tile block.
// A 16x32: lane->M; lanes<16 hold K{0-7,16-23}, lanes>=16 K{8-15,24-31}.
// B 32x16: lane->N; lanes<16 hold K{0-7,16-23}, lanes>=16 K{8-15,24-31}.
template<int LDS_S>
__device__ __forceinline__ void wmma_block(const bf16* __restrict__ As,
                                           const bf16* __restrict__ Bs,
                                           v8f (&acc)[2][4],
                                           int wm, int wn, int l15, int hsel) {
    v16bf afrag[2], bfrag[4];
#pragma unroll
    for (int i = 0; i < 2; ++i) {
        const bf16* ar = &As[(wm + 16 * i + l15) * LDS_S];
        v8bf lo = *(const v8bf*)(ar + hsel * 8);
        v8bf hi = *(const v8bf*)(ar + 16 + hsel * 8);
#pragma unroll
        for (int e = 0; e < 8; ++e) { afrag[i][e] = lo[e]; afrag[i][8 + e] = hi[e]; }
    }
#pragma unroll
    for (int j = 0; j < 4; ++j) {
        const bf16* br = &Bs[(wn + 16 * j + l15) * LDS_S];
        v8bf lo = *(const v8bf*)(br + hsel * 8);
        v8bf hi = *(const v8bf*)(br + 16 + hsel * 8);
#pragma unroll
        for (int e = 0; e < 8; ++e) { bfrag[j][e] = lo[e]; bfrag[j][8 + e] = hi[e]; }
    }
#pragma unroll
    for (int i = 0; i < 2; ++i)
#pragma unroll
        for (int j = 0; j < 4; ++j)
            acc[i][j] = __builtin_amdgcn_wmma_f32_16x16x32_bf16(
                false, afrag[i], false, bfrag[j], (short)0, acc[i][j],
                false, false);
}

// ---------------------------------------------------------------------------
// Tiled WMMA GEMM on bf16 operands: C[M,N] = op(A) @ op(B) (+bias)(+relu)
// OPA: 0 = A row-major [M,K]; 1 = logical transpose of row-major [K,M].
// OPB: 1 = B row-major [N,K] (B^T operand, contiguous K rows);
//      0 = B row-major [K,N].
// ascale (optional, len K): logical A[m,k] *= ascale[k] (trilinear x*w3).
// Dual outputs Cf (fp32) / Cbf (bf16); either may be null.
// Macrotile 128x128, K step 32; 8 waves, each 32x64 (2x4 wmma tiles).
// Requires M%128==0, N%128==0, K%32==0 (true for every call site).
// When OPA==0 && OPB==1 && !ascale, A/B tiles are TDM-pipelined (2 LDS bufs).
// ---------------------------------------------------------------------------
template<int OPA, int OPB>
__global__ __launch_bounds__(256)
void gemm_wmma_bf16(const bf16* __restrict__ A, long strideA, int lda,
                    const bf16* __restrict__ B, long strideB, int ldb,
                    float* __restrict__ Cf, bf16* __restrict__ Cbf,
                    long strideC, int ldc,
                    const float* __restrict__ bias,
                    const float* __restrict__ ascale,
                    int M, int N, int K, int relu)
{
    constexpr int MT = 128, NTL = 128, KT = 32, LDS_S = KT + 8;
    constexpr int NBUF = (OPA == 0 && OPB == 1 && HAS_TDM) ? 2 : 1;
    __shared__ bf16 As[NBUF * MT * LDS_S];   // As[buf][m][k]
    __shared__ bf16 Bs[NBUF * NTL * LDS_S];  // Bs[buf][n][k]

    const int tid = threadIdx.x;
    const int bz  = blockIdx.z;
    const bf16* Ab = A + (long)bz * strideA;
    const bf16* Bb = B + (long)bz * strideB;

    const int m0 = blockIdx.y * MT;
    const int n0 = blockIdx.x * NTL;

    const int wave = tid >> 5;
    const int lane = tid & 31;
    const int wm   = (wave & 3) * 32;   // 4 waves along M
    const int wn   = (wave >> 2) * 64;  // 2 waves along N
    const int l15  = lane & 15;
    const int hsel = lane >> 4;

    v8f acc[2][4] = {};
    bool handled = false;

#if HAS_TDM
    if constexpr (OPA == 0 && OPB == 1) {
        if (ascale == nullptr) {
            // ---- double-buffered TDM pipeline: DMA(i+1) overlaps WMMA(i) ----
            const int nk = K / KT;
            if (tid < 32) {
                tdm_load_tile(lds_off_u32(&As[0]), Ab + (long)m0 * lda,
                              KT, MT, (unsigned)lda);
                tdm_load_tile(lds_off_u32(&Bs[0]), Bb + (long)n0 * ldb,
                              KT, NTL, (unsigned)ldb);
            }
            for (int i = 0; i < nk; ++i) {
                const int cur = i & 1;
                if (tid < 32) {
                    if (i + 1 < nk) {
                        const int nxt = (i + 1) & 1;
                        const long kn = (long)(i + 1) * KT;
                        tdm_load_tile(lds_off_u32(&As[nxt * MT * LDS_S]),
                                      Ab + (long)m0 * lda + kn, KT, MT,
                                      (unsigned)lda);
                        tdm_load_tile(lds_off_u32(&Bs[nxt * NTL * LDS_S]),
                                      Bb + (long)n0 * ldb + kn, KT, NTL,
                                      (unsigned)ldb);
                        // tile i's 2 loads retired; tile i+1 (2) stays in flight
                        __builtin_amdgcn_s_wait_tensorcnt(2);
                    } else {
                        __builtin_amdgcn_s_wait_tensorcnt(0);
                    }
                }
                __syncthreads();
                wmma_block<LDS_S>(&As[cur * MT * LDS_S], &Bs[cur * NTL * LDS_S],
                                  acc, wm, wn, l15, hsel);
                __syncthreads();  // buffer consumed before tile i+2 overwrites
            }
            handled = true;
        }
    }
#endif

    if (!handled) {
        for (int k0 = 0; k0 < K; k0 += KT) {
            __syncthreads();  // previous tile fully consumed

            // B tile first (TDM overlaps the VALU A-staging below)
            if constexpr (OPB == 1) {  // phys [N,K]: contiguous K rows
#if HAS_TDM
                if (tid < 32)
                    tdm_load_tile(lds_off_u32(&Bs[0]),
                                  Bb + (long)n0 * ldb + k0, KT, NTL,
                                  (unsigned)ldb);
#else
                if (tid == 0 && k0 + KT < K)
                    __builtin_prefetch(Bb + (long)n0 * ldb + (k0 + KT), 0, 1);
                for (int c = tid; c < (NTL * KT) / 8; c += 256) {
                    int n = c >> 2, ko = (c & 3) * 8;
                    *(v8bf*)&Bs[n * LDS_S + ko] =
                        *(const v8bf*)&Bb[(long)(n0 + n) * ldb + k0 + ko];
                }
#endif
            } else {  // phys [K,N]
                if (tid == 0 && k0 + KT < K)
                    __builtin_prefetch(Bb + (long)(k0 + KT) * ldb + n0, 0, 1);
                for (int c = tid; c < (KT * NTL) / 8; c += 256) {
                    int n8 = c & 15, k = c >> 4;
                    v8bf v = *(const v8bf*)&Bb[(long)(k0 + k) * ldb + n0 + n8 * 8];
#pragma unroll
                    for (int e = 0; e < 8; ++e)
                        Bs[(n8 * 8 + e) * LDS_S + k] = v[e];
                }
            }

            // A tile
            if constexpr (OPA == 0) {
                if (ascale) {
                    for (int idx = tid; idx < MT * KT; idx += 256) {
                        int m = idx >> 5, k = idx & 31;
                        As[m * LDS_S + k] =
                            f2bf(bf2f(Ab[(long)(m0 + m) * lda + k0 + k]) *
                                 ascale[k0 + k]);
                    }
                } else {
                    for (int c = tid; c < (MT * KT) / 8; c += 256) {
                        int m = c >> 2, ko = (c & 3) * 8;
                        *(v8bf*)&As[m * LDS_S + ko] =
                            *(const v8bf*)&Ab[(long)(m0 + m) * lda + k0 + ko];
                    }
                }
            } else {  // OPA == 1: logical A[m,k] = phys[k*lda + m]
                for (int c = tid; c < (MT * KT) / 8; c += 256) {
                    int m8 = c & 15, k = c >> 4;
                    v8bf v = *(const v8bf*)&Ab[(long)(k0 + k) * lda + m0 + m8 * 8];
#pragma unroll
                    for (int e = 0; e < 8; ++e)
                        As[(m8 * 8 + e) * LDS_S + k] = v[e];
                }
            }

#if HAS_TDM
            if (OPB == 1 && tid < 32) __builtin_amdgcn_s_wait_tensorcnt(0);
#endif
            __syncthreads();
            wmma_block<LDS_S>(&As[0], &Bs[0], acc, wm, wn, l15, hsel);
        }
    }

    // ---- epilogue: C/D layout lane->N, VGPR r -> M=r (+8 for lanes 16-31) ----
#pragma unroll
    for (int i = 0; i < 2; ++i) {
#pragma unroll
        for (int j = 0; j < 4; ++j) {
            int n  = n0 + wn + 16 * j + l15;
            int mb = m0 + wm + 16 * i + hsel * 8;
            float bv = bias ? bias[n] : 0.0f;
#pragma unroll
            for (int r = 0; r < 8; ++r) {
                float v = acc[i][j][r] + bv;
                if (relu) v = fmaxf(v, 0.0f);
                long off = (long)bz * strideC + (long)(mb + r) * ldc + n;
                if (Cf)  Cf[off]  = v;
                if (Cbf) Cbf[off] = f2bf(v);
            }
        }
    }
}

// ---------------------------------------------------------------------------
// Small kernels
// ---------------------------------------------------------------------------
__global__ __launch_bounds__(256)
void cvt_kernel(const float* __restrict__ in, bf16* __restrict__ out) {
    long idx = (long)blockIdx.x * 256 + threadIdx.x;
    out[idx] = f2bf(in[idx]);
}

// out[c*R + r] = in[r*C + c]  (fp32 [R,C] -> bf16 [C,R]); one-time per weight
__global__ __launch_bounds__(256)
void cvt_t_kernel(const float* __restrict__ in, bf16* __restrict__ out,
                  int R, int C) {
    long idx = (long)blockIdx.x * 256 + threadIdx.x;
    long r = idx / C;
    int  c = (int)(idx % C);
    out[(long)c * R + r] = f2bf(in[idx]);
}

// trip[bt, 0:384] = concat(E[r[bt,0]], E[r[bt,1]], E[r[bt,2]]) -> bf16
__global__ __launch_bounds__(256)
void gather_trip_kernel(const int* __restrict__ r, const float* __restrict__ E,
                        bf16* __restrict__ trip) {
    long idx = (long)blockIdx.x * 256 + threadIdx.x;
    long bt = idx / K3;
    int  c  = (int)(idx % K3);
    int  e  = c >> 7, off = c & 127;
    trip[idx] = f2bf(E[(long)r[bt * 3 + e] * TEE + off]);
}

// part1[row] = dot(X[row,:], w); one wave per row (wave32)
__global__ __launch_bounds__(256)
void rowdot_kernel(const bf16* __restrict__ X, const float* __restrict__ w,
                   float* __restrict__ out, int cols) {
    int row  = blockIdx.x * 8 + (threadIdx.x >> 5);
    int lane = threadIdx.x & 31;
    const bf16* xr = X + (long)row * cols;
    float s = 0.0f;
    for (int c = lane; c < cols; c += 32) s += bf2f(xr[c]) * w[c];
#pragma unroll
    for (int off = 16; off; off >>= 1) s += __shfl_xor(s, off, 32);
    if (lane == 0) out[row] = s;
}

// softmax over T of A[b][t][s] (+ part1[b*T+t]); writes bf16 att.
// part2 / b1 / b2 are constant along T and cancel.
__global__ __launch_bounds__(256)
void softmax_t_kernel(float* __restrict__ A, const float* __restrict__ p1,
                      bf16* __restrict__ att) {
    int b = blockIdx.y;
    int s = blockIdx.x * blockDim.x + threadIdx.x;
    float* Ab = A + (long)b * NT * NS;
    bf16*  Ob = att + (long)b * NT * NS;
    const float* pb = p1 + (long)b * NT;
    float mx = -3.4e38f;
    for (int t = 0; t < NT; ++t)
        mx = fmaxf(mx, Ab[(long)t * NS + s] + pb[t]);
    float sum = 0.0f;
    for (int t = 0; t < NT; ++t) {
        float v = __expf(Ab[(long)t * NS + s] + pb[t] - mx);
        sum += v;
        Ab[(long)t * NS + s] = v;
    }
    float inv = 1.0f / sum;
    for (int t = 0; t < NT; ++t)
        Ob[(long)t * NS + s] = f2bf(Ab[(long)t * NS + s] * inv);
}

// cat[bs, off+h] = ok - olk ; cat[bs, off+H+h] = ok * olk  (bf16 out)
__global__ __launch_bounds__(256)
void rule_cat_kernel(const float* __restrict__ ok, const float* __restrict__ olk,
                     bf16* __restrict__ cat, int ruleOff) {
    long idx = (long)blockIdx.x * 256 + threadIdx.x;  // BT*H
    long bs = idx / HH;
    int  h  = (int)(idx % HH);
    float o = ok[idx], l = olk[idx];
    bf16* base = cat + bs * (12 * HH) + ruleOff;
    base[h]      = f2bf(o - l);
    base[HH + h] = f2bf(o * l);
}

// OFK = concat([G, OSK, G-OSK, G*OSK]) (bf16 out)
__global__ __launch_bounds__(256)
void ofk_kernel(const float* __restrict__ G, const float* __restrict__ OSK,
                bf16* __restrict__ OFK) {
    long idx = (long)blockIdx.x * 256 + threadIdx.x;  // BT*H
    long bs = idx / HH;
    int  h  = (int)(idx % HH);
    float g = G[idx], o = OSK[idx];
    bf16* base = OFK + bs * (4 * HH);
    base[h]          = f2bf(g);
    base[HH + h]     = f2bf(o);
    base[2 * HH + h] = f2bf(g - o);
    base[3 * HH + h] = f2bf(g * o);
}

// ---------------------------------------------------------------------------
// Host-side dispatch
// ---------------------------------------------------------------------------
static inline void run_gemm(int opa,
                            const bf16* A, long sA, int lda,
                            const bf16* B, long sB, int ldb,
                            float* Cf, bf16* Cbf, long sC, int ldc,
                            const float* bias, const float* ascale,
                            int M, int N, int K, int relu, int batch,
                            hipStream_t st) {
    dim3 g(N / 128, M / 128, batch), blk(256);
    if (opa == 0)  // A row-major, B^T layout
        gemm_wmma_bf16<0, 1><<<g, blk, 0, st>>>(A, sA, lda, B, sB, ldb,
                                                Cf, Cbf, sC, ldc, bias, ascale,
                                                M, N, K, relu);
    else           // A transposed, B row-major [K,N]
        gemm_wmma_bf16<1, 0><<<g, blk, 0, st>>>(A, sA, lda, B, sB, ldb,
                                                Cf, Cbf, sC, ldc, bias, ascale,
                                                M, N, K, relu);
}

extern "C" void kernel_launch(void* const* d_in, const int* in_sizes, int n_in,
                              void* d_out, int out_size, void* d_ws, size_t ws_size,
                              hipStream_t stream)
{
    const int* r_enc[6];
    for (int j = 0; j < 6; ++j) r_enc[j] = (const int*)d_in[j];
    const float* ok   = (const float*)d_in[6];   // attention_output [B,S,H]
    // d_in[7] roberta_output: feeds only the discarded co-attention branch.
    const float* E    = (const float*)d_in[8];
    const float* mlpW = (const float*)d_in[9];
    const float* mlpb = (const float*)d_in[10];
    const float* liiW = (const float*)d_in[11];
    const float* rw1  = (const float*)d_in[12];
    const float* rw3  = (const float*)d_in[16];
    const float* wc   = (const float*)d_in[17];
    const float* bc   = (const float*)d_in[18];
    const float* wg   = (const float*)d_in[19];
    const float* bg   = (const float*)d_in[20];
    const float* sw1  = (const float*)d_in[28];
    const float* sw3  = (const float*)d_in[32];
    const float* wf   = (const float*)d_in[33];
    const float* bfv  = (const float*)d_in[34];
    float* out = (float*)d_out;
    (void)in_sizes; (void)n_in; (void)out_size; (void)ws_size;

    // -------- workspace carve (256B aligned) --------
    char* wp = (char*)d_ws;
    auto carve = [&](long bytes) {
        char* p = wp; wp += (bytes + 255) & ~255L; return p;
    };
    bf16* tripb = (bf16*)carve((long)BT * K3 * 2);
    bf16* X1b   = (bf16*)carve((long)BT * K3 * 2);
    bf16* olb   = (bf16*)carve((long)BT * HH * 2);
    bf16* attb  = (bf16*)carve((long)NB * NT * NS * 2);
    bf16* catb  = (bf16*)carve((long)BT * 12 * HH * 2);
    bf16* Yb    = (bf16*)carve((long)BT * 6 * HH * 2);
    bf16* Gb    = (bf16*)carve((long)BT * HH * 2);
    bf16* OFKb  = (bf16*)carve((long)BT * 4 * HH * 2);
    bf16* okb   = (bf16*)carve((long)BT * HH * 2);
    bf16* mlpWT = (bf16*)carve((long)K3 * K3 * 2);          // [384,384] (N,K)
    bf16* liiWT = (bf16*)carve((long)HH * K3 * 2);          // [1024,384]
    bf16* wcT   = (bf16*)carve((long)6 * HH * 12 * HH * 2); // [6144,12288]
    bf16* wgT   = (bf16*)carve((long)HH * 6 * HH * 2);      // [1024,6144]
    bf16* wfT   = (bf16*)carve((long)HH * 4 * HH * 2);      // [1024,4096]
    float* p1   = (float*)carve((long)BT * 4);
    float* Abuf = (float*)carve((long)NB * NT * NS * 4);
    float* olkf = (float*)carve((long)BT * HH * 4);
    float* Gf   = (float*)carve((long)BT * HH * 4);
    float* OSKf = (float*)carve((long)BT * HH * 4);

    const long sTH = (long)NT * HH;  // per-batch [512,1024]
    const long sTS = (long)NT * NS;  // per-batch [512,512]

    // -------- one-time conversions: weights -> transposed bf16, ok -> bf16 --
    cvt_t_kernel<<<(long)K3 * K3 / 256, 256, 0, stream>>>(mlpW, mlpWT, K3, K3);
    cvt_t_kernel<<<(long)K3 * HH / 256, 256, 0, stream>>>(liiW, liiWT, K3, HH);
    cvt_t_kernel<<<(long)12 * HH * 6 * HH / 256, 256, 0, stream>>>(wc, wcT, 12 * HH, 6 * HH);
    cvt_t_kernel<<<(long)6 * HH * HH / 256, 256, 0, stream>>>(wg, wgT, 6 * HH, HH);
    cvt_t_kernel<<<(long)4 * HH * HH / 256, 256, 0, stream>>>(wf, wfT, 4 * HH, HH);
    cvt_kernel<<<(long)BT * HH / 256, 256, 0, stream>>>(ok, okb);

    // -------- per-rule pipeline --------
    for (int j = 0; j < 6; ++j) {
        gather_trip_kernel<<<(BT * K3) / 256, 256, 0, stream>>>(r_enc[j], E, tripb);
        // X1 = trip @ mlp_W + mlp_b            [4096,384]
        run_gemm(0, tripb, 0, K3, mlpWT, 0, K3, nullptr, X1b, 0, K3,
                 mlpb, nullptr, BT, K3, K3, 0, 1, stream);
        // ol = X1 @ lii_W                      [4096,1024]
        run_gemm(0, X1b, 0, K3, liiWT, 0, K3, nullptr, olb, 0, HH,
                 nullptr, nullptr, BT, HH, K3, 0, 1, stream);
        // part1 = ol @ rule_w1 (b1 cancels in the softmax)
        rowdot_kernel<<<BT / 8, 256, 0, stream>>>(olb, rw1, p1, HH);
        // P3[b] = (ol*w3)[b] @ ok[b]^T         [8][512,512]
        run_gemm(0, olb, sTH, HH, okb, sTH, HH, Abuf, nullptr, sTS, NS,
                 nullptr, rw3, NT, NS, HH, 0, NB, stream);
        // att = softmax_T(P3 + part1) -> bf16
        softmax_t_kernel<<<dim3(NS / 256, NB), 256, 0, stream>>>(Abuf, p1, attb);
        // olk[b] = att[b]^T @ ol[b]            [8][512,1024]
        run_gemm(1, attb, sTS, NS, olb, sTH, HH, olkf, nullptr, sTH, HH,
                 nullptr, nullptr, NS, HH, NT, 0, NB, stream);
        // cat[:, j*2H:(j+1)*2H] = [ok - olk, ok * olk]
        rule_cat_kernel<<<(long)BT * HH / 256, 256, 0, stream>>>(ok, olkf, catb,
                                                                 j * 2 * HH);
    }

    // -------- merge --------
    // Y = cat @ wc + bc   [4096,6144]   (TDM-pipelined weight GEMM)
    run_gemm(0, catb, 0, 12 * HH, wcT, 0, 12 * HH, nullptr, Yb, 0, 6 * HH,
             bc, nullptr, BT, 6 * HH, 12 * HH, 0, 1, stream);
    // G = Y @ wg + bg     [4096,1024] (dual fp32 + bf16 outputs)
    run_gemm(0, Yb, 0, 6 * HH, wgT, 0, 6 * HH, Gf, Gb, 0, HH,
             bg, nullptr, BT, HH, 6 * HH, 0, 1, stream);

    // -------- self-attention --------
    rowdot_kernel<<<BT / 8, 256, 0, stream>>>(Gb, sw1, p1, HH);
    // A2[b] = (G*sa_w3)[b] @ G[b]^T
    run_gemm(0, Gb, sTH, HH, Gb, sTH, HH, Abuf, nullptr, sTS, NS,
             nullptr, sw3, NT, NS, HH, 0, NB, stream);
    softmax_t_kernel<<<dim3(NS / 256, NB), 256, 0, stream>>>(Abuf, p1, attb);
    // OSK[b] = att2[b]^T @ G[b]
    run_gemm(1, attb, sTS, NS, Gb, sTH, HH, OSKf, nullptr, sTH, HH,
             nullptr, nullptr, NS, HH, NT, 0, NB, stream);
    // OFK = [G, OSK, G-OSK, G*OSK]
    ofk_kernel<<<(long)BT * HH / 256, 256, 0, stream>>>(Gf, OSKf, OFKb);
    // out = relu(OFK @ wf + bf)
    run_gemm(0, OFKb, 0, 4 * HH, wfT, 0, 4 * HH, out, nullptr, 0, HH,
             bfv, nullptr, BT, HH, 4 * HH, 1, 1, stream);
}
